// AttnDecoder_52647709114845
// MI455X (gfx1250) — matrix operand order repaired and checked by reading, not compiled, not run on hardware
//
#include <hip/hip_runtime.h>
#include <math.h>

// Shapes (fixed by the reference)
#define VOCAB 32000
#define DIM   512
#define HID   1024
#define AKD   100      // attention key dim
#define BB    32
#define SS    128
#define TTOT  20
#define TM1   19

typedef __attribute__((ext_vector_type(16))) __bf16 v16bf;
typedef __attribute__((ext_vector_type(8)))  float  v8f;

union Frag { v16bf v; uint4 q[2]; };

__device__ __forceinline__ unsigned short f2bf(float f) {
  union { float f; unsigned int u; } x; x.f = f;
  unsigned int u = x.u;
  unsigned int r = u + 0x7FFFu + ((u >> 16) & 1u);   // round-to-nearest-even
  return (unsigned short)(r >> 16);
}
__device__ __forceinline__ float sigf(float x) { return 1.0f / (1.0f + __expf(-x)); }

// A fragment (16x32 bf16, M x K), row-major activation source.
// lanes 0-15: M=lane, K = k0+0..7 / k0+16..23 ; lanes 16-31: M=lane-16, K = +8 / +24
__device__ __forceinline__ v16bf load_a_frag(const unsigned short* A, int lda,
                                             int m0, int k0, int lane) {
  int m = m0 + (lane & 15);
  int k = k0 + ((lane >> 4) << 3);
  const unsigned short* p = A + (size_t)m * lda + k;
  Frag f;
  f.q[0] = *(const uint4*)(p);        // K .. K+7
  f.q[1] = *(const uint4*)(p + 16);   // K+16 .. K+23
  return f.v;
}

// B fragment (32x16 bf16, K x N), pre-swizzled: per (ntile,ktile,lane) 16 contiguous bf16
__device__ __forceinline__ v16bf load_b_frag(const unsigned short* Bp, int Ktiles,
                                             int nt, int kt, int lane) {
  const unsigned short* p = Bp + ((((size_t)nt * Ktiles) + kt) * 32 + lane) * 16;
  Frag f;
  f.q[0] = *(const uint4*)(p);
  f.q[1] = *(const uint4*)(p + 8);
  return f.v;
}

__device__ __forceinline__ v8f wmma_bf16(v16bf a, v16bf b, v8f c) {
  return __builtin_amdgcn_wmma_f32_16x16x32_bf16(false, a, false, b, (short)0, c,
                                                 false, false);
}

// ---------------- weight pack: W (N,K) f32 row-major -> B-fragment bf16 tiles
__global__ void pack_b_kernel(const float* __restrict__ W, unsigned short* __restrict__ out,
                              int N, int K, int Ktiles, int total) {
  int gid = blockIdx.x * blockDim.x + threadIdx.x;
  if (gid >= total) return;
  int lane = gid & 31;
  int tile = gid >> 5;
  int kt = tile % Ktiles;
  int nt = tile / Ktiles;
  int n  = nt * 16 + (lane & 15);
  int kb = kt * 32 + ((lane >> 4) << 3);
  unsigned short* o = out + (size_t)gid * 16;
#pragma unroll
  for (int j = 0; j < 16; ++j) {
    int k = kb + (j < 8 ? j : j + 8);
    float v = (n < N && k < K) ? W[(size_t)n * K + k] : 0.0f;
    o[j] = f2bf(v);
  }
}

// pack two stacked weights: columns [0,K1) from W1 (N,K1), [K1,K1+K2) from W2 (N,K2)
__global__ void pack_b2_kernel(const float* __restrict__ W1, int K1,
                               const float* __restrict__ W2, int K2,
                               unsigned short* __restrict__ out,
                               int Ktiles, int total) {
  int gid = blockIdx.x * blockDim.x + threadIdx.x;
  if (gid >= total) return;
  int lane = gid & 31;
  int tile = gid >> 5;
  int kt = tile % Ktiles;
  int nt = tile / Ktiles;
  int n  = nt * 16 + (lane & 15);
  int kb = kt * 32 + ((lane >> 4) << 3);
  unsigned short* o = out + (size_t)gid * 16;
#pragma unroll
  for (int j = 0; j < 16; ++j) {
    int k = kb + (j < 8 ? j : j + 8);
    float v = (k < K1) ? W1[(size_t)n * K1 + k] : W2[(size_t)n * K2 + (k - K1)];
    o[j] = f2bf(v);
  }
}

__global__ void vec_add_kernel(const float* a, const float* b, float* o, int n) {
  int i = blockIdx.x * blockDim.x + threadIdx.x;
  if (i < n) o[i] = a[i] + b[i];
}
__global__ void zero_f32_kernel(float* p, int n) {
  int i = blockIdx.x * blockDim.x + threadIdx.x;
  if (i < n) p[i] = 0.0f;
}
__global__ void zero_u32_kernel(unsigned int* p, int n) {
  int i = blockIdx.x * blockDim.x + threadIdx.x;
  if (i < n) p[i] = 0u;
}
__global__ void cvt_bf16_kernel(const float* in, unsigned short* out, int n) {
  int i = blockIdx.x * blockDim.x + threadIdx.x;
  if (i < n) out[i] = f2bf(in[i]);
}

// embedding gathers (fp32 table -> bf16 activations)
__global__ void embed_src_kernel(const float* __restrict__ emb, const int* __restrict__ seq,
                                 unsigned short* __restrict__ out, int total) {
  int i = blockIdx.x * blockDim.x + threadIdx.x;       // over B*S*D
  if (i >= total) return;
  int d  = i % DIM;
  int bs = i / DIM;                                    // b*S+s
  int tok = seq[bs];
  out[i] = f2bf(emb[(size_t)tok * DIM + d]);
}
__global__ void embed_dec_kernel(const float* __restrict__ emb, const int* __restrict__ trg,
                                 unsigned short* __restrict__ out, int total) {
  int i = blockIdx.x * blockDim.x + threadIdx.x;       // over TM1*B*D
  if (i >= total) return;
  int d  = i % DIM;
  int tb = i / DIM;
  int b  = tb % BB;
  int t  = tb / BB;                                    // 0..TM1-1
  int tok = trg[b * TTOT + t];
  out[i] = f2bf(emb[(size_t)tok * DIM + d]);
}

// ---------------- generic WMMA GEMM: C[M,N] = A[M,K] @ B + bias, act, store f32/bf16
// grid.x = M/32, grid.y = Ntiles/NTW, block = 32 (one wave). NTW N-tiles per wave
// amortize each A fragment over NTW WMMAs (cuts A-side L2 traffic NTW x).
template <int NTW>
__global__ void gemm_kernel(const unsigned short* __restrict__ A, int lda,
                            const unsigned short* __restrict__ Bp, int Ktiles,
                            const float* __restrict__ bias,
                            float* Cf, int ldcf,
                            unsigned short* Cbf, int ldch,
                            int Nreal, int act) {
  int lane = threadIdx.x & 31;
  int m0 = blockIdx.x * 32;
  int ntBase = blockIdx.y * NTW;
  v8f acc0[NTW], acc1[NTW];
#pragma unroll
  for (int j = 0; j < NTW; ++j) {
    int n = (ntBase + j) * 16 + (lane & 15);
    float bv = bias[n < Nreal ? n : 0];
#pragma unroll
    for (int r = 0; r < 8; ++r) { acc0[j][r] = bv; acc1[j][r] = bv; }
  }
  for (int kt = 0; kt < Ktiles; ++kt) {
    if (kt + 1 < Ktiles)
      __builtin_prefetch(Bp + ((((size_t)ntBase * Ktiles) + kt + 1) * 32 + lane) * 16, 0, 3);
    v16bf a0 = load_a_frag(A, lda, m0,      kt * 32, lane);
    v16bf a1 = load_a_frag(A, lda, m0 + 16, kt * 32, lane);
#pragma unroll
    for (int j = 0; j < NTW; ++j) {
      v16bf bf = load_b_frag(Bp, Ktiles, ntBase + j, kt, lane);
      acc0[j] = wmma_bf16(a0, bf, acc0[j]);
      acc1[j] = wmma_bf16(a1, bf, acc1[j]);
    }
  }
#pragma unroll
  for (int j = 0; j < NTW; ++j) {
    int n = (ntBase + j) * 16 + (lane & 15);
    if (n >= Nreal) continue;
#pragma unroll
    for (int r = 0; r < 8; ++r) {
      int row0 = m0 + ((lane >> 4) << 3) + r;
      int row1 = row0 + 16;
      float v0 = acc0[j][r], v1 = acc1[j][r];
      if (act == 1) { v0 = fmaxf(v0, 0.0f); v1 = fmaxf(v1, 0.0f); }
      else if (act == 2) { v0 = tanhf(v0); v1 = tanhf(v1); }
      if (Cf)  { Cf[(size_t)row0 * ldcf + n] = v0;        Cf[(size_t)row1 * ldcf + n] = v1; }
      if (Cbf) { Cbf[(size_t)row0 * ldch + n] = f2bf(v0); Cbf[(size_t)row1 * ldch + n] = f2bf(v1); }
    }
  }
}

// ---------------- fused LSTM step, latency-optimized for the sequential chain:
// grid = H/16 blocks, block = 256 threads = 8 waves.
// wave w: gate g = w&3, K-half ks = w>>2  -> each wave does 2 WMMAs per k-iter over
// half of K. The K range is pre-split into up to 3 tight sub-loops (x | ctx | h
// activation sources, all boundaries are tile-aligned) so the inner loops carry no
// divergent source selection. Partial gate pre-activations are combined in LDS,
// then the LSTM pointwise (sigmoid/tanh/mask) runs across all 256 threads.
__global__ void __launch_bounds__(256)
lstm_step_kernel(const unsigned short* __restrict__ A1, int ld1, int K1,
                 const unsigned short* __restrict__ A2, int ld2, int K2,
                 const unsigned short* __restrict__ A3, int ld3,
                 int Ktiles,
                 const unsigned short* __restrict__ Wp,  // packed (4H x K)
                 const float* __restrict__ bias,         // 4H
                 float* __restrict__ c,                  // [B,H] in/out
                 float* __restrict__ hf,                 // [B,H] in/out
                 unsigned short* __restrict__ hbf,       // [B,H] out (next buf)
                 unsigned short* __restrict__ hbf2, int ld2b, // extra bf16 out
                 const int* __restrict__ lens, int t) {
  __shared__ float zb[2][4][BB * 16];   // [ksplit][gate][b*16+nl]  16 KB
  int tid  = threadIdx.x;
  int lane = tid & 31;
  int w    = tid >> 5;
  int g    = w & 3;
  int ks   = w >> 2;
  int nt   = blockIdx.x;              // hidden cols [nt*16, nt*16+16)
  int ktH  = Ktiles >> 1;
  int kt0  = ks * ktH;
  int ktE  = kt0 + ktH;
  int btile = g * (HID / 16) + nt;    // gate's N tile

  v8f acc0, acc1;
#pragma unroll
  for (int r = 0; r < 8; ++r) { acc0[r] = 0.0f; acc1[r] = 0.0f; }

  // per-source tile ranges (uniform, tile-aligned)
  int t1 = K1 >> 5;            // tiles [0, t1)           -> A1
  int t2 = (K1 + K2) >> 5;     // tiles [t1, t2)          -> A2
                               // tiles [t2, Ktiles)      -> A3
  const unsigned short* srcA[3] = { A1, A2, A3 };
  int srcLd[3]  = { ld1, ld2, ld3 };
  int srcLo[3]  = { 0, t1, t2 };
  int srcHi[3]  = { t1, t2, Ktiles };
#pragma unroll
  for (int sIdx = 0; sIdx < 3; ++sIdx) {
    const unsigned short* Ak = srcA[sIdx];
    int ldk = srcLd[sIdx];
    int lo = kt0 > srcLo[sIdx] ? kt0 : srcLo[sIdx];
    int hi = ktE < srcHi[sIdx] ? ktE : srcHi[sIdx];
    int kbase = srcLo[sIdx] * 32;
    for (int kt = lo; kt < hi; ++kt) {
      v16bf a0 = load_a_frag(Ak, ldk, 0,  kt * 32 - kbase, lane);
      v16bf a1 = load_a_frag(Ak, ldk, 16, kt * 32 - kbase, lane);
      v16bf bf = load_b_frag(Wp, Ktiles, btile, kt, lane);
      acc0 = wmma_bf16(a0, bf, acc0);
      acc1 = wmma_bf16(a1, bf, acc1);
    }
  }

  // spill partial z to LDS (accumulator layout -> [b][nl])
#pragma unroll
  for (int r = 0; r < 8; ++r) {
    int b0 = ((lane >> 4) << 3) + r;
    zb[ks][g][b0 * 16 + (lane & 15)]        = acc0[r];
    zb[ks][g][(b0 + 16) * 16 + (lane & 15)] = acc1[r];
  }
  __syncthreads();

  // pointwise LSTM: 512 elements over 256 threads
#pragma unroll
  for (int e = tid; e < BB * 16; e += 256) {
    int b  = e >> 4;
    int nl = e & 15;
    int n  = nt * 16 + nl;
    float zi = zb[0][0][e] + zb[1][0][e] + bias[0 * HID + n];
    float zf = zb[0][1][e] + zb[1][1][e] + bias[1 * HID + n];
    float zg = zb[0][2][e] + zb[1][2][e] + bias[2 * HID + n];
    float zo = zb[0][3][e] + zb[1][3][e] + bias[3 * HID + n];
    size_t idx = (size_t)b * HID + n;
    float cold = c[idx];
    float cn = sigf(zf) * cold + sigf(zi) * tanhf(zg);
    float hn = sigf(zo) * tanhf(cn);
    bool msk = lens ? (t < lens[b]) : true;
    float hold = hf[idx];
    float hsel = msk ? hn : hold;
    float csel = msk ? cn : cold;
    c[idx] = csel;
    hf[idx] = hsel;
    hbf[idx] = f2bf(hsel);
    if (hbf2) {
      float ov = lens ? (msk ? hn : 0.0f) : hsel;
      hbf2[(size_t)b * ld2b + n] = f2bf(ov);
    }
  }
}

// ---------------- attention core: energy dot + masked softmax + context
// grid = B blocks, block = S(=128) threads
__global__ void attn_core_kernel(const float* __restrict__ qkey,  // [B,S,AKD]
                                 const float* __restrict__ akey,  // [B,AKD]
                                 const float* __restrict__ qval,  // [B,S,HID]
                                 const int* __restrict__ lens,
                                 unsigned short* __restrict__ ctx_bf,  // [B,HID]
                                 unsigned short* __restrict__ hc_cat)  // [B,2H], writes [H,2H)
{
  __shared__ float red[SS];
  __shared__ float w[SS];
  int b = blockIdx.x;
  int s = threadIdx.x;
  float e = 0.0f;
  const float* qk = qkey + ((size_t)b * SS + s) * AKD;
  const float* ak = akey + (size_t)b * AKD;
  for (int k = 0; k < AKD; ++k) e += qk[k] * ak[k];
  bool valid = s < lens[b];
  e = valid ? e : -1e30f;
  red[s] = e; __syncthreads();
  for (int off = SS / 2; off > 0; off >>= 1) {
    if (s < off) red[s] = fmaxf(red[s], red[s + off]);
    __syncthreads();
  }
  float mx = red[0]; __syncthreads();
  float ex = valid ? __expf(e - mx) : 0.0f;
  red[s] = ex; __syncthreads();
  for (int off = SS / 2; off > 0; off >>= 1) {
    if (s < off) red[s] += red[s + off];
    __syncthreads();
  }
  float inv = 1.0f / red[0];
  w[s] = ex * inv;
  __syncthreads();
  // context: each thread does 8 hidden cols
#pragma unroll
  for (int j = 0; j < HID / SS; ++j) {
    int col = j * SS + s;
    float acc = 0.0f;
    for (int ss = 0; ss < SS; ++ss)
      acc += w[ss] * qval[((size_t)b * SS + ss) * HID + col];
    unsigned short v = f2bf(acc);
    ctx_bf[(size_t)b * HID + col] = v;
    hc_cat[(size_t)b * 2 * HID + HID + col] = v;
  }
}

// =====================================================================
extern "C" void kernel_launch(void* const* d_in, const int* in_sizes, int n_in,
                              void* d_out, int out_size, void* d_ws, size_t ws_size,
                              hipStream_t stream) {
  const float* embed    = (const float*)d_in[0];
  const float* enc_Wih  = (const float*)d_in[1];
  const float* enc_Whh  = (const float*)d_in[2];
  const float* enc_bih  = (const float*)d_in[3];
  const float* enc_bhh  = (const float*)d_in[4];
  const float* dec_Wih  = (const float*)d_in[5];
  const float* dec_Whh  = (const float*)d_in[6];
  const float* dec_bih  = (const float*)d_in[7];
  const float* dec_bhh  = (const float*)d_in[8];
  const float* qk_W     = (const float*)d_in[9];
  const float* qk_b     = (const float*)d_in[10];
  const float* qv_W     = (const float*)d_in[11];
  const float* qv_b     = (const float*)d_in[12];
  const float* ak_W     = (const float*)d_in[13];
  const float* ak_b     = (const float*)d_in[14];
  const float* out_W    = (const float*)d_in[15];
  const float* out_b    = (const float*)d_in[16];
  const float* wd_b     = (const float*)d_in[17];
  const float* hfc1_W   = (const float*)d_in[18];
  const float* hfc1_b   = (const float*)d_in[19];
  const float* hfc2_W   = (const float*)d_in[20];
  const float* hfc2_b   = (const float*)d_in[21];
  const float* cfc1_W   = (const float*)d_in[22];
  const float* cfc1_b   = (const float*)d_in[23];
  const float* cfc2_W   = (const float*)d_in[24];
  const float* cfc2_b   = (const float*)d_in[25];
  const int*   src_seqs = (const int*)d_in[26];
  const int*   src_lens = (const int*)d_in[27];
  const int*   trg_seqs = (const int*)d_in[28];
  float* out = (float*)d_out;   // [B, TM1, V]

  // ---- workspace carve-out
  char* base = (char*)d_ws;
  size_t off = 0;
  auto alloc = [&](size_t bytes) -> void* {
    void* p = base + off;
    off = (off + bytes + 255) & ~(size_t)255;
    return p;
  };
  // packed bf16 weights (B-fragment layout)
  unsigned short* pEmb = (unsigned short*)alloc((size_t)2000 * 16 * 512 * 2);
  unsigned short* pEnc = (unsigned short*)alloc((size_t)256 * 48 * 512 * 2);
  unsigned short* pDec = (unsigned short*)alloc((size_t)256 * 80 * 512 * 2);
  unsigned short* pQK  = (unsigned short*)alloc((size_t)7 * 32 * 512 * 2);
  unsigned short* pQV  = (unsigned short*)alloc((size_t)64 * 32 * 512 * 2);
  unsigned short* pAK  = (unsigned short*)alloc((size_t)7 * 32 * 512 * 2);
  unsigned short* pOUT = (unsigned short*)alloc((size_t)32 * 64 * 512 * 2);
  unsigned short* pH1  = (unsigned short*)alloc((size_t)128 * 32 * 512 * 2);
  unsigned short* pH2  = (unsigned short*)alloc((size_t)64 * 64 * 512 * 2);
  unsigned short* pC1  = (unsigned short*)alloc((size_t)128 * 32 * 512 * 2);
  unsigned short* pC2  = (unsigned short*)alloc((size_t)64 * 64 * 512 * 2);
  // activations
  unsigned short* Xsrc   = (unsigned short*)alloc((size_t)BB * SS * DIM * 2);
  unsigned short* Xdec   = (unsigned short*)alloc((size_t)TM1 * BB * DIM * 2);
  unsigned short* srcHid = (unsigned short*)alloc((size_t)BB * SS * HID * 2);
  float* qKey  = (float*)alloc((size_t)BB * SS * AKD * 4);
  float* qVal  = (float*)alloc((size_t)BB * SS * HID * 4);
  float* hEnc  = (float*)alloc((size_t)BB * HID * 4);
  float* cEnc  = (float*)alloc((size_t)BB * HID * 4);
  unsigned short* hEncBf = (unsigned short*)alloc((size_t)2 * BB * HID * 2); // ping-pong
  unsigned short* cEncBf = (unsigned short*)alloc((size_t)BB * HID * 2);
  float* biasE = (float*)alloc(4096 * 4);
  float* biasD = (float*)alloc(4096 * 4);
  unsigned short* tmpBf  = (unsigned short*)alloc((size_t)BB * 2 * HID * 2);
  float* hDec  = (float*)alloc((size_t)BB * HID * 4);
  float* cDec  = (float*)alloc((size_t)BB * HID * 4);
  unsigned short* hDecBf = (unsigned short*)alloc((size_t)2 * BB * HID * 2); // ping-pong
  float* aKey  = (float*)alloc((size_t)BB * AKD * 4);
  unsigned short* ctxBf  = (unsigned short*)alloc((size_t)BB * HID * 2);
  unsigned short* hcCat  = (unsigned short*)alloc((size_t)BB * 2 * HID * 2);
  unsigned short* featBf = (unsigned short*)alloc((size_t)BB * DIM * 2);
  (void)ws_size; (void)in_sizes; (void)n_in; (void)out_size;

  const int TB = 256;
  auto blks = [](int total, int tb) { return (total + tb - 1) / tb; };

  // ---- pack weights to bf16 B-fragment tiles
  { int tot = 2000 * 16 * 32;
    pack_b_kernel<<<blks(tot, TB), TB, 0, stream>>>(embed, pEmb, VOCAB, DIM, 16, tot); }
  { int tot = 256 * 48 * 32;
    pack_b2_kernel<<<blks(tot, TB), TB, 0, stream>>>(enc_Wih, DIM, enc_Whh, HID, pEnc, 48, tot); }
  { int tot = 256 * 80 * 32;
    pack_b2_kernel<<<blks(tot, TB), TB, 0, stream>>>(dec_Wih, DIM + HID, dec_Whh, HID, pDec, 80, tot); }
  { int tot = 7 * 32 * 32;
    pack_b_kernel<<<blks(tot, TB), TB, 0, stream>>>(qk_W, pQK, AKD, HID, 32, tot);
    pack_b_kernel<<<blks(tot, TB), TB, 0, stream>>>(ak_W, pAK, AKD, HID, 32, tot); }
  { int tot = 64 * 32 * 32;
    pack_b_kernel<<<blks(tot, TB), TB, 0, stream>>>(qv_W, pQV, HID, HID, 32, tot); }
  { int tot = 32 * 64 * 32;
    pack_b_kernel<<<blks(tot, TB), TB, 0, stream>>>(out_W, pOUT, DIM, 2 * HID, 64, tot); }
  { int tot = 128 * 32 * 32;
    pack_b_kernel<<<blks(tot, TB), TB, 0, stream>>>(hfc1_W, pH1, 2 * HID, HID, 32, tot);
    pack_b_kernel<<<blks(tot, TB), TB, 0, stream>>>(cfc1_W, pC1, 2 * HID, HID, 32, tot); }
  { int tot = 64 * 64 * 32;
    pack_b_kernel<<<blks(tot, TB), TB, 0, stream>>>(hfc2_W, pH2, HID, 2 * HID, 64, tot);
    pack_b_kernel<<<blks(tot, TB), TB, 0, stream>>>(cfc2_W, pC2, HID, 2 * HID, 64, tot); }

  vec_add_kernel<<<blks(4096, TB), TB, 0, stream>>>(enc_bih, enc_bhh, biasE, 4096);
  vec_add_kernel<<<blks(4096, TB), TB, 0, stream>>>(dec_bih, dec_bhh, biasD, 4096);

  { int tot = BB * SS * DIM;
    embed_src_kernel<<<blks(tot, TB), TB, 0, stream>>>(embed, src_seqs, Xsrc, tot); }
  { int tot = TM1 * BB * DIM;
    embed_dec_kernel<<<blks(tot, TB), TB, 0, stream>>>(embed, trg_seqs, Xdec, tot); }

  zero_f32_kernel<<<blks(BB * HID, TB), TB, 0, stream>>>(hEnc, BB * HID);
  zero_f32_kernel<<<blks(BB * HID, TB), TB, 0, stream>>>(cEnc, BB * HID);
  zero_u32_kernel<<<blks(BB * HID, TB), TB, 0, stream>>>((unsigned int*)hEncBf, BB * HID); // both bufs

  // ---- encoder: 128 sequential fused LSTM steps (8 waves/block: gate x K-split)
  int cur = 0;
  for (int t = 0; t < SS; ++t) {
    int nxt = cur ^ 1;
    lstm_step_kernel<<<HID / 16, 256, 0, stream>>>(
        Xsrc + (size_t)t * DIM, SS * DIM, DIM,          // x_t  (cols 0..511)
        hEncBf + (size_t)cur * BB * HID, HID, HID,      // h    (cols 512..1535)
        hEncBf + (size_t)cur * BB * HID, HID,           // (unused third source)
        48, pEnc, biasE, cEnc, hEnc,
        hEncBf + (size_t)nxt * BB * HID,
        srcHid + (size_t)t * HID, SS * HID,             // masked enc_out slice
        src_lens, t);
    cur = nxt;
  }
  unsigned short* hFinBf = hEncBf + (size_t)cur * BB * HID;
  cvt_bf16_kernel<<<blks(BB * HID, TB), TB, 0, stream>>>(cEnc, cEncBf, BB * HID);

  // ---- attention precompute over src_hidden (M = B*S = 4096)
  { dim3 g(BB * SS / 32, 7);
    gemm_kernel<1><<<g, 32, 0, stream>>>(srcHid, HID, pQK, 32, qk_b,
                                         qKey, AKD, nullptr, 0, AKD, 2); }      // tanh
  { dim3 g(BB * SS / 32, 64 / 4);
    gemm_kernel<4><<<g, 32, 0, stream>>>(srcHid, HID, pQV, 32, qv_b,
                                         qVal, HID, nullptr, 0, HID, 0); }

  // ---- bridge MLPs: dh, dc
  { dim3 g(1, 128 / 4);
    gemm_kernel<4><<<g, 32, 0, stream>>>(hFinBf, HID, pH1, 32, hfc1_b,
                                         nullptr, 0, tmpBf, 2 * HID, 2 * HID, 1); } // relu
  { dim3 g(1, 64 / 4);
    gemm_kernel<4><<<g, 32, 0, stream>>>(tmpBf, 2 * HID, pH2, 64, hfc2_b,
                                         hDec, HID, hDecBf, HID, HID, 0); }
  { dim3 g(1, 128 / 4);
    gemm_kernel<4><<<g, 32, 0, stream>>>(cEncBf, HID, pC1, 32, cfc1_b,
                                         nullptr, 0, tmpBf, 2 * HID, 2 * HID, 1); }
  { dim3 g(1, 64 / 4);
    gemm_kernel<4><<<g, 32, 0, stream>>>(tmpBf, 2 * HID, pC2, 64, cfc2_b,
                                         cDec, HID, nullptr, 0, HID, 0); }

  // ---- decoder: 19 sequential steps (attention + fused LSTM + out + logits)
  int dcur = 0;
  for (int t = 0; t < TM1; ++t) {
    int dnxt = dcur ^ 1;
    unsigned short* hCur = hDecBf + (size_t)dcur * BB * HID;
    // a_key = tanh(h @ ak_W^T + ak_b)
    { dim3 g(1, 7);
      gemm_kernel<1><<<g, 32, 0, stream>>>(hCur, HID, pAK, 32, ak_b,
                                           aKey, AKD, nullptr, 0, AKD, 2); }
    // energies + softmax + context (writes ctxBf and hcCat[:, H:2H))
    attn_core_kernel<<<BB, SS, 0, stream>>>(qKey, aKey, qVal, src_lens, ctxBf, hcCat);
    // LSTM step on [x_t | context | h]
    lstm_step_kernel<<<HID / 16, 256, 0, stream>>>(
        Xdec + (size_t)t * BB * DIM, DIM, DIM,   // x_t   (cols 0..511)
        ctxBf, HID, HID,                         // ctx   (cols 512..1535)
        hCur, HID,                               // h     (cols 1536..2559)
        80, pDec, biasD, cDec, hDec,
        hDecBf + (size_t)dnxt * BB * HID,
        hcCat, 2 * HID,                          // also place h_new into hc_cat[:, 0:H)
        nullptr, t);
    // feat = [h_new, context] @ out_W^T + out_b  (bf16)
    { dim3 g(1, 32 / 4);
      gemm_kernel<4><<<g, 32, 0, stream>>>(hcCat, 2 * HID, pOUT, 64, out_b,
                                           nullptr, 0, featBf, DIM, DIM, 0); }
    // logits = feat @ embed^T + wd_b  -> out[b][t][:]
    { dim3 g(1, VOCAB / 16 / 4);
      gemm_kernel<4><<<g, 32, 0, stream>>>(featBf, DIM, pEmb, 16, wd_b,
                                           out + (size_t)t * VOCAB, TM1 * VOCAB,
                                           nullptr, 0, VOCAB, 0); }
    dcur = dnxt;
  }
}
